// PatchTST_25254407700637
// MI455X (gfx1250) — matrix operand order repaired and checked by reading, not compiled, not run on hardware
//
#include <hip/hip_runtime.h>
#include <math.h>
#include <stdint.h>

// ---- problem constants (from reference) ----
#define Bsz     32
#define LIN     336
#define Mch     64
#define Pp      16
#define STRIDEp 4
#define Nn      81          // patches
#define Dd      512
#define Hh      8
#define Ee      64          // head dim
#define DFFd    1024
#define NLl     3
#define BMn     (Bsz * Mch)     // 2048 sequences
#define Ttok    (BMn * Nn)      // 165888 tokens (multiple of 128)
#define EPSf    1e-5f

typedef __attribute__((ext_vector_type(16))) _Float16 v16h;
typedef __attribute__((ext_vector_type(8)))  _Float16 v8h;
typedef __attribute__((ext_vector_type(8)))  float    v8f;

// ---------------------------------------------------------------------------
// gfx1250 async copy: global -> LDS, 16B per lane, tracked with ASYNCcnt.
// VDST = per-lane LDS byte address (low 32 bits of the flat shared pointer,
// per the LDS aperture rule addr[31:0] == LDS offset), VADDR = 64-bit global.
// ---------------------------------------------------------------------------
__device__ __forceinline__ void async_ld_b128(const _Float16* g, _Float16* l) {
  uint32_t lds = (uint32_t)(uintptr_t)(void*)l;
  uint64_t ga  = (uint64_t)(uintptr_t)(const void*)g;
  asm volatile("global_load_async_to_lds_b128 %0, %1, off"
               :: "v"(lds), "v"(ga) : "memory");
}
__device__ __forceinline__ void wait_async0() {
  asm volatile("s_wait_asynccnt 0" ::: "memory");
}

// ---------------------------------------------------------------------------
// fp32 -> fp16 conversion (weights staged once per launch)
// ---------------------------------------------------------------------------
__global__ __launch_bounds__(256)
void cvt_f32_f16(const float* __restrict__ s, _Float16* __restrict__ d, int n) {
  int i = blockIdx.x * 256 + threadIdx.x;
  if (i < n) d[i] = (_Float16)s[i];
}

__global__ __launch_bounds__(256)
void zero_f32(float* __restrict__ p, int n) {
  int i = blockIdx.x * 256 + threadIdx.x;
  if (i < n) p[i] = 0.0f;
}

// ---------------------------------------------------------------------------
// Patchify + circular Conv1d(P->D, k=3) + sinusoidal pos-emb.
// ---------------------------------------------------------------------------
__global__ __launch_bounds__(256)
void patch_embed(const float* __restrict__ x, const float* __restrict__ tokw,
                 float* __restrict__ h32, _Float16* __restrict__ h16) {
  const int bm = blockIdx.x;           // 0..2047
  const int b = bm / Mch, m = bm % Mch;
  __shared__ float patch[Nn][Pp];
  for (int idx = threadIdx.x; idx < Nn * Pp; idx += 256) {
    int n = idx / Pp, p = idx % Pp;
    patch[n][p] = x[((size_t)b * LIN + (n * STRIDEp + p)) * Mch + m];
  }
  __syncthreads();
  const float lnw = logf(10000.0f);
  for (int dpass = 0; dpass < 2; ++dpass) {
    const int d = threadIdx.x + dpass * 256;
    float w[48];
#pragma unroll
    for (int p = 0; p < Pp; ++p)
#pragma unroll
      for (int kk = 0; kk < 3; ++kk)
        w[kk * Pp + p] = tokw[((size_t)d * Pp + p) * 3 + kk];
    const int i2 = (d >> 1) * 2;
    const float div = __expf(-lnw * (float)i2 / (float)Dd);
    for (int n = 0; n < Nn; ++n) {
      float acc = 0.f;
#pragma unroll
      for (int kk = 0; kk < 3; ++kk) {
        int nn = n + kk - 1;
        nn = (nn < 0) ? nn + Nn : (nn >= Nn ? nn - Nn : nn);
#pragma unroll
        for (int p = 0; p < Pp; ++p) acc += patch[nn][p] * w[kk * Pp + p];
      }
      const float ang = (float)n * div;
      acc += (d & 1) ? __cosf(ang) : __sinf(ang);
      const size_t o = ((size_t)bm * Nn + n) * Dd + d;
      h32[o] = acc;
      h16[o] = (_Float16)acc;
    }
  }
}

// ---------------------------------------------------------------------------
// WMMA GEMM:  C[t,o] = act( sum_k A[t,k] * W[o,k] + bias[o] )
// A: [T,K] fp16 row-major, W: [Dout,K] fp16 row-major (B^T form).
// Block tile 128x128, K staged 64 at a time through LDS via async-to-LDS
// copies. 8 waves in a 4x2 grid; each wave owns 32x64 = 2x4 WMMA tiles,
// so 16 v_wmma_f32_16x16x32_f16 per stage between one pair of barriers.
// Fragment gathers follow ISA 7.12.2 wave32 layouts:
//   A lane<16: K{0..7,16..23}; lane>=16: K{8..15,24..31}  (two b128 LDS loads)
//   B lane<16: K{0..7,16..23}; lane>=16: K{8..15,24..31}  (two b128 LDS loads)
// ---------------------------------------------------------------------------
__global__ __launch_bounds__(256)
void gemm_wmma(const _Float16* __restrict__ A, const _Float16* __restrict__ W,
               const float* __restrict__ bias, float* C32, _Float16* C16,
               int K, int Dout, int act) {
  __shared__ _Float16 As[128][72];   // 64 halves + 8 pad (144B stride)
  __shared__ _Float16 Bs[128][72];
  const int tid = threadIdx.x;
  const int lane = tid & 31, wave = tid >> 5;
  const int waveM = (wave & 3) * 32;   // 4 waves down
  const int waveN = (wave >> 2) * 64;  // 2 waves across
  const int rowBase = blockIdx.x * 128;
  const int colBase = blockIdx.y * 128;
  const _Float16* Ag = A + (size_t)rowBase * K;
  const _Float16* Wg = W + (size_t)colBase * K;

  v8f acc[2][4] = {};

  for (int kc = 0; kc < K; kc += 64) {
    // async-stage A and B tiles: 128 rows x 64 halves each, 16B per issue
#pragma unroll
    for (int c = tid; c < 128 * 8; c += 256) {
      const int r = c >> 3, s = c & 7;
      async_ld_b128(Ag + (size_t)r * K + kc + s * 8, &As[r][s * 8]);
      async_ld_b128(Wg + (size_t)r * K + kc + s * 8, &Bs[r][s * 8]);
      if (kc + 64 < K)  // keep a global_prefetch_b8 for the next stage
        __builtin_prefetch(Ag + (size_t)r * K + kc + 64 + s * 8, 0, 1);
    }
    wait_async0();
    __syncthreads();

#pragma unroll
    for (int ks = 0; ks < 64; ks += 32) {
      const int am = waveM + (lane & 15);
      const int akb = ks + ((lane < 16) ? 0 : 8);
      v8h alo0 = *reinterpret_cast<const v8h*>(&As[am][akb]);
      v8h ahi0 = *reinterpret_cast<const v8h*>(&As[am][akb + 16]);
      v8h alo1 = *reinterpret_cast<const v8h*>(&As[am + 16][akb]);
      v8h ahi1 = *reinterpret_cast<const v8h*>(&As[am + 16][akb + 16]);
      v16h a0 = __builtin_shufflevector(alo0, ahi0, 0, 1, 2, 3, 4, 5, 6, 7, 8,
                                        9, 10, 11, 12, 13, 14, 15);
      v16h a1 = __builtin_shufflevector(alo1, ahi1, 0, 1, 2, 3, 4, 5, 6, 7, 8,
                                        9, 10, 11, 12, 13, 14, 15);
      const int bn = waveN + (lane & 15);
      const int bkb = ks + ((lane < 16) ? 0 : 8);
      v16h bf[4];
#pragma unroll
      for (int nj = 0; nj < 4; ++nj) {
        v8h lo = *reinterpret_cast<const v8h*>(&Bs[bn + nj * 16][bkb]);
        v8h hi = *reinterpret_cast<const v8h*>(&Bs[bn + nj * 16][bkb + 16]);
        bf[nj] = __builtin_shufflevector(lo, hi, 0, 1, 2, 3, 4, 5, 6, 7, 8, 9,
                                         10, 11, 12, 13, 14, 15);
      }
#pragma unroll
      for (int nj = 0; nj < 4; ++nj) {
        acc[0][nj] = __builtin_amdgcn_wmma_f32_16x16x32_f16(
            false, a0, false, bf[nj], (short)0, acc[0][nj], false, false);
        acc[1][nj] = __builtin_amdgcn_wmma_f32_16x16x32_f16(
            false, a1, false, bf[nj], (short)0, acc[1][nj], false, false);
      }
    }
    __syncthreads();
  }

  // epilogue: C fragment layout (ISA 7.12.2): VGPR r, lanes 0-15 -> M=r,
  // lanes 16-31 -> M=r+8; N = lane&15.
#pragma unroll
  for (int mi = 0; mi < 2; ++mi)
#pragma unroll
    for (int nj = 0; nj < 4; ++nj) {
      const int col = colBase + waveN + nj * 16 + (lane & 15);
      const float bcol = bias ? bias[col] : 0.0f;
#pragma unroll
      for (int r = 0; r < 8; ++r) {
        const int row =
            rowBase + waveM + mi * 16 + ((lane < 16) ? 0 : 8) + r;
        float v = acc[mi][nj][r] + bcol;
        if (act) v = 0.5f * v * (1.0f + erff(v * 0.70710678118654752f));
        const size_t o = (size_t)row * Dout + col;
        if (C32) C32[o] = v;
        if (C16) C16[o] = (_Float16)v;
      }
    }
}

// ---------------------------------------------------------------------------
// Attention: one block per (sequence, head). K_h and V_h live in LDS (fp32),
// each wave owns query rows i = wave, wave+8, ... Softmax over 81 keys via
// wave32 shuffles; explicit s_wait_dscnt for the in-wave LDS p-row handoff.
// ---------------------------------------------------------------------------
__global__ __launch_bounds__(256)
void attention(const _Float16* __restrict__ q, const _Float16* __restrict__ k,
               const _Float16* __restrict__ v, _Float16* __restrict__ o) {
  const int bm = blockIdx.x, h = blockIdx.y;
  const int tid = threadIdx.x, lane = tid & 31, wave = tid >> 5;
  __shared__ float Ks[Nn][Ee + 1];
  __shared__ float Vs[Nn][Ee + 1];
  __shared__ float Pr[8][Nn + 3];
  const size_t hb = (size_t)bm * Nn * Dd + (size_t)h * Ee;
  for (int idx = tid; idx < Nn * Ee; idx += 256) {
    const int j = idx >> 6, e = idx & 63;
    const size_t g = hb + (size_t)j * Dd + e;
    Ks[j][e] = (float)k[g];
    Vs[j][e] = (float)v[g];
  }
  __syncthreads();
  const float scale = 0.125f;  // 1/sqrt(64)
  for (int i = wave; i < Nn; i += 8) {
    const _Float16* qi = q + hb + (size_t)i * Dd;
    const int j0 = lane, j1 = lane + 32, j2 = lane + 64;
    float s0 = 0.f, s1 = 0.f, s2 = 0.f;
    for (int e = 0; e < Ee; ++e) {
      const float qe = (float)qi[e];
      s0 += qe * Ks[j0][e];
      s1 += qe * Ks[j1][e];
      if (j2 < Nn) s2 += qe * Ks[j2][e];
    }
    s0 *= scale; s1 *= scale; s2 *= scale;
    float mx = fmaxf(s0, s1);
    if (j2 < Nn) mx = fmaxf(mx, s2);
    for (int off = 16; off > 0; off >>= 1)
      mx = fmaxf(mx, __shfl_xor(mx, off, 32));
    const float p0 = __expf(s0 - mx);
    const float p1 = __expf(s1 - mx);
    const float p2 = (j2 < Nn) ? __expf(s2 - mx) : 0.f;
    float sum = p0 + p1 + p2;
    for (int off = 16; off > 0; off >>= 1) sum += __shfl_xor(sum, off, 32);
    const float inv = 1.0f / sum;
    Pr[wave][j0] = p0;
    Pr[wave][j1] = p1;
    if (j2 < Nn) Pr[wave][j2] = p2;
    asm volatile("s_wait_dscnt 0" ::: "memory");  // in-wave LDS RAW
    for (int e = lane; e < Ee; e += 32) {
      float acc = 0.f;
      for (int j = 0; j < Nn; ++j) acc += Pr[wave][j] * Vs[j][e];
      o[hb + (size_t)i * Dd + e] = (_Float16)(acc * inv);
    }
  }
}

// ---------------------------------------------------------------------------
// BatchNorm over (BM*N) rows per channel on (a + b).
// ---------------------------------------------------------------------------
__global__ __launch_bounds__(256)
void bn_stats(const float* __restrict__ a, const float* __restrict__ b,
              float* __restrict__ sum, float* __restrict__ sumsq) {
  const int row0 = blockIdx.x * 64;
  const int c0 = threadIdx.x, c1 = threadIdx.x + 256;
  float s0 = 0.f, s1 = 0.f, q0 = 0.f, q1 = 0.f;
  for (int r = 0; r < 64; ++r) {
    const size_t base = (size_t)(row0 + r) * Dd;
    const float va = a[base + c0] + b[base + c0];
    const float vb = a[base + c1] + b[base + c1];
    s0 += va; q0 += va * va;
    s1 += vb; q1 += vb * vb;
  }
  atomicAdd(&sum[c0], s0);
  atomicAdd(&sum[c1], s1);
  atomicAdd(&sumsq[c0], q0);
  atomicAdd(&sumsq[c1], q1);
}

__global__ __launch_bounds__(256)
void bn_norm(const float* a, const float* __restrict__ b,
             const float* __restrict__ sum, const float* __restrict__ sumsq,
             const float* __restrict__ g, const float* __restrict__ bb,
             float* h32, _Float16* __restrict__ h16, float* out) {
  const float invT = 1.0f / (float)Ttok;
  const size_t total = (size_t)Ttok * Dd;
  size_t i = (size_t)blockIdx.x * 256 + threadIdx.x;
  const size_t step = (size_t)gridDim.x * 256;
  for (; i < total; i += step) {
    const int c = (int)(i & (Dd - 1));
    const float mean = sum[c] * invT;
    const float var = sumsq[c] * invT - mean * mean;
    const float y =
        (a[i] + b[i] - mean) * rsqrtf(var + EPSf) * g[c] + bb[c];
    h32[i] = y;
    h16[i] = (_Float16)y;
    if (out) out[i] = y;
  }
}

// ---------------------------------------------------------------------------
// Orchestration
// ---------------------------------------------------------------------------
extern "C" void kernel_launch(void* const* d_in, const int* in_sizes, int n_in,
                              void* d_out, int out_size, void* d_ws,
                              size_t ws_size, hipStream_t stream) {
  (void)in_sizes; (void)n_in; (void)out_size; (void)ws_size;
  const float* x_enc = (const float*)d_in[0];
  const float* tokw  = (const float*)d_in[1];
  const float* wq = (const float*)d_in[2];
  const float* bq = (const float*)d_in[3];
  const float* wk = (const float*)d_in[4];
  const float* bk = (const float*)d_in[5];
  const float* wv = (const float*)d_in[6];
  const float* bv = (const float*)d_in[7];
  const float* wo = (const float*)d_in[8];
  const float* bo = (const float*)d_in[9];
  const float* c1w = (const float*)d_in[10];
  const float* c1b = (const float*)d_in[11];
  const float* c2w = (const float*)d_in[12];
  const float* c2b = (const float*)d_in[13];
  const float* g1  = (const float*)d_in[14];
  const float* be1 = (const float*)d_in[15];
  const float* g2  = (const float*)d_in[16];
  const float* be2 = (const float*)d_in[17];

  char* ws = (char*)d_ws;
  size_t off = 0;
  auto alloc = [&](size_t bytes) -> void* {
    void* p = ws + off;
    off = (off + bytes + 255) & ~(size_t)255;
    return p;
  };
  const size_t nDD = (size_t)NLl * Dd * Dd;    // 786432
  const size_t nFD = (size_t)NLl * DFFd * Dd;  // 1572864
  _Float16* wq16 = (_Float16*)alloc(nDD * 2);
  _Float16* wk16 = (_Float16*)alloc(nDD * 2);
  _Float16* wv16 = (_Float16*)alloc(nDD * 2);
  _Float16* wo16 = (_Float16*)alloc(nDD * 2);
  _Float16* c1w16 = (_Float16*)alloc(nFD * 2);
  _Float16* c2w16 = (_Float16*)alloc(nFD * 2);
  float*    h32 = (float*)alloc((size_t)Ttok * Dd * 4);
  _Float16* h16 = (_Float16*)alloc((size_t)Ttok * Dd * 2);
  _Float16* q16 = (_Float16*)alloc((size_t)Ttok * Dd * 2);
  _Float16* k16 = (_Float16*)alloc((size_t)Ttok * Dd * 2);
  _Float16* v16 = (_Float16*)alloc((size_t)Ttok * Dd * 2);
  _Float16* o16 = (_Float16*)alloc((size_t)Ttok * Dd * 2);
  _Float16* y116 = (_Float16*)alloc((size_t)Ttok * DFFd * 2);
  float*    r32 = (float*)alloc((size_t)Ttok * Dd * 4);
  float*    stats = (float*)alloc(1024 * 4);  // [sum(512), sumsq(512)]

  // weight staging
  cvt_f32_f16<<<(int)((nDD + 255) / 256), 256, 0, stream>>>(wq, wq16, (int)nDD);
  cvt_f32_f16<<<(int)((nDD + 255) / 256), 256, 0, stream>>>(wk, wk16, (int)nDD);
  cvt_f32_f16<<<(int)((nDD + 255) / 256), 256, 0, stream>>>(wv, wv16, (int)nDD);
  cvt_f32_f16<<<(int)((nDD + 255) / 256), 256, 0, stream>>>(wo, wo16, (int)nDD);
  cvt_f32_f16<<<(int)((nFD + 255) / 256), 256, 0, stream>>>(c1w, c1w16, (int)nFD);
  cvt_f32_f16<<<(int)((nFD + 255) / 256), 256, 0, stream>>>(c2w, c2w16, (int)nFD);

  // embed + pos
  patch_embed<<<BMn, 256, 0, stream>>>(x_enc, tokw, h32, h16);

  const dim3 gD(Ttok / 128, Dd / 128);    // 1296 x 4
  const dim3 gF(Ttok / 128, DFFd / 128);  // 1296 x 8
  const int statsBlocks = Ttok / 64;      // 2592
  const int normBlocks = 20736;           // grid-stride over 84.9M elems

  for (int l = 0; l < NLl; ++l) {
    const _Float16* Wq = wq16 + (size_t)l * Dd * Dd;
    const _Float16* Wk = wk16 + (size_t)l * Dd * Dd;
    const _Float16* Wv = wv16 + (size_t)l * Dd * Dd;
    const _Float16* Wo = wo16 + (size_t)l * Dd * Dd;
    const _Float16* W1 = c1w16 + (size_t)l * DFFd * Dd;
    const _Float16* W2 = c2w16 + (size_t)l * Dd * DFFd;

    gemm_wmma<<<gD, 256, 0, stream>>>(h16, Wq, bq + l * Dd, nullptr, q16, Dd, Dd, 0);
    gemm_wmma<<<gD, 256, 0, stream>>>(h16, Wk, bk + l * Dd, nullptr, k16, Dd, Dd, 0);
    gemm_wmma<<<gD, 256, 0, stream>>>(h16, Wv, bv + l * Dd, nullptr, v16, Dd, Dd, 0);
    attention<<<dim3(BMn, Hh), 256, 0, stream>>>(q16, k16, v16, o16);
    gemm_wmma<<<gD, 256, 0, stream>>>(o16, Wo, bo + l * Dd, r32, nullptr, Dd, Dd, 0);

    zero_f32<<<4, 256, 0, stream>>>(stats, 1024);
    bn_stats<<<statsBlocks, 256, 0, stream>>>(h32, r32, stats, stats + 512);
    bn_norm<<<normBlocks, 256, 0, stream>>>(h32, r32, stats, stats + 512,
                                            g1 + l * Dd, be1 + l * Dd,
                                            h32, h16, nullptr);

    gemm_wmma<<<gF, 256, 0, stream>>>(h16, W1, c1b + l * DFFd, nullptr, y116, Dd, DFFd, 1);
    gemm_wmma<<<gD, 256, 0, stream>>>(y116, W2, c2b + l * Dd, r32, nullptr, DFFd, Dd, 0);

    zero_f32<<<4, 256, 0, stream>>>(stats, 1024);
    bn_stats<<<statsBlocks, 256, 0, stream>>>(h32, r32, stats, stats + 512);
    bn_norm<<<normBlocks, 256, 0, stream>>>(h32, r32, stats, stats + 512,
                                            g2 + l * Dd, be2 + l * Dd, h32, h16,
                                            (l == NLl - 1) ? (float*)d_out
                                                           : nullptr);
  }
}
